// GestureBiLSTM_10651518894519
// MI455X (gfx1250) — matrix operand order repaired and checked by reading, not compiled, not run on hardware
//
#include <hip/hip_runtime.h>
#include <hip/hip_bf16.h>
#include <stdint.h>

// ---------------------------------------------------------------------------
// GestureBiLSTM for MI455X (gfx1250): persistent fused LSTM with bf16 WMMA.
// B=1024, T=256, D=63, H=128, O=14.
// ---------------------------------------------------------------------------

#define BSZ   1024
#define TSTEP 256
#define DD    63
#define HH    128

typedef __bf16 bf16x16 __attribute__((ext_vector_type(16)));
typedef float  f32x8   __attribute__((ext_vector_type(8)));

struct Frag32 { uint4 lo, hi; };

__device__ __forceinline__ bf16x16 as_frag(uint4 a, uint4 b) {
    Frag32 f{a, b};
    return __builtin_bit_cast(bf16x16, f);
}

__device__ __forceinline__ unsigned int pack_bf16(float lo, float hi) {
    unsigned int ul = __builtin_bit_cast(unsigned int, lo);
    unsigned int uh = __builtin_bit_cast(unsigned int, hi);
    ul = (ul + 0x7fffu + ((ul >> 16) & 1u)) >> 16;   // RNE
    uh = (uh + 0x7fffu + ((uh >> 16) & 1u)) >> 16;
    return ul | (uh << 16);
}

__device__ __forceinline__ unsigned short f2bf(float f) {
    unsigned int u = __builtin_bit_cast(unsigned int, f);
    u = (u + 0x7fffu + ((u >> 16) & 1u)) >> 16;
    return (unsigned short)u;
}

// Gate nonlinearities on the serial critical path: use hardware-rate
// v_rcp_f32 / v_exp_f32 (and v_tanh_f32 if available) instead of IEEE
// division sequences (v_div_scale/v_div_fmas/v_div_fixup).
__device__ __forceinline__ float sigmoid_(float v) {
    return __builtin_amdgcn_rcpf(1.0f + __expf(-v));
}
__device__ __forceinline__ float tanh_(float v) {
#if __has_builtin(__builtin_amdgcn_tanhf)
    return __builtin_amdgcn_tanhf(v);
#else
    return 1.0f - 2.0f * __builtin_amdgcn_rcpf(1.0f + __expf(2.0f * v));
#endif
}

// ---------------------------------------------------------------------------
// Prep weights: fp32 [4H][K] -> bf16 pairs in [n][k/2] dword layout,
// K zero-padded 63->64 (W_ih), 128 (W_hh).  bsum = b_ih + b_hh.
// ---------------------------------------------------------------------------
__global__ __launch_bounds__(256)
void prep_kernel(const float* __restrict__ Wih, const float* __restrict__ Whh,
                 const float* __restrict__ bih, const float* __restrict__ bhh,
                 unsigned int* __restrict__ wihq,   // 512*32 dwords
                 unsigned int* __restrict__ whhq,   // 512*64 dwords
                 float* __restrict__ bsum)          // 512
{
    int idx = blockIdx.x * 256 + threadIdx.x;
    if (idx < 512 * 32) {
        int n = idx >> 5, kd = idx & 31, k = kd * 2;
        float lo = (k     < DD) ? Wih[n * DD + k]     : 0.f;
        float hi = (k + 1 < DD) ? Wih[n * DD + k + 1] : 0.f;
        wihq[idx] = pack_bf16(lo, hi);
    } else if (idx < 512 * 32 + 512 * 64) {
        int j = idx - 512 * 32;
        int n = j >> 6, kd = j & 63, k = kd * 2;
        whhq[j] = pack_bf16(Whh[n * HH + k], Whh[n * HH + k + 1]);
    } else if (idx < 512 * 32 + 512 * 64 + 512) {
        int j = idx - (512 * 32 + 512 * 64);
        bsum[j] = bih[j] + bhh[j];
    }
}

// ---------------------------------------------------------------------------
// Prep x: fp32 [B][T][63] -> bf16 pairs pre-swizzled into per-(btile,t) 2KB
// blocks laid out exactly in WMMA A-fragment dword order:
//   xbf[bt][t][m*32 + k/2]   (m = row in 16-row tile, k padded 63->64)
// ---------------------------------------------------------------------------
__global__ __launch_bounds__(256)
void prep_x_kernel(const float* __restrict__ x, unsigned int* __restrict__ xbf)
{
    size_t idx = (size_t)blockIdx.x * 256 + threadIdx.x;   // < 64*256*512
    int kd = (int)(idx & 31);
    int m  = (int)((idx >> 5) & 15);
    int t  = (int)((idx >> 9) & 255);
    int bt = (int)(idx >> 17);
    const float* xr = x + ((size_t)(bt * 16 + m) * TSTEP + t) * DD;
    int k = kd * 2;
    float lo = (k     < DD) ? xr[k]     : 0.f;
    float hi = (k + 1 < DD) ? xr[k + 1] : 0.f;
    xbf[idx] = pack_bf16(lo, hi);
}

// ---------------------------------------------------------------------------
// Fused persistent LSTM direction kernel pieces
// ---------------------------------------------------------------------------
__device__ __forceinline__ void proj_wmma(f32x8 acc[4], const unsigned int* sXr,
                                          const uint4 wA[4][2], const uint4 wB[4][2],
                                          int nn, int half)
{
#pragma unroll
    for (int kt = 0; kt < 2; ++kt) {
        int da = nn * 32 + kt * 16 + half * 4;
        bf16x16 a = as_frag(*(const uint4*)(sXr + da),
                            *(const uint4*)(sXr + da + 8));
#pragma unroll
        for (int g = 0; g < 4; ++g) {
            acc[g] = __builtin_amdgcn_wmma_f32_16x16x32_bf16(
                false, a, false, as_frag(wA[g][kt], wB[g][kt]),
                (short)0, acc[g], false, false);
        }
    }
}

__device__ __forceinline__ void rec_wmma(f32x8 acc[4], const unsigned int* sHr,
                                         const uint4 hA[4][2], const uint4 hB[4][2],
                                         const uint4* sWhh, int wave, int nn, int half)
{
    // kt=0,1: B-fragments pinned in registers
#pragma unroll
    for (int kt = 0; kt < 2; ++kt) {
        int da = nn * 64 + kt * 16 + half * 4;
        bf16x16 a = as_frag(*(const uint4*)(sHr + da),
                            *(const uint4*)(sHr + da + 8));
#pragma unroll
        for (int g = 0; g < 4; ++g) {
            acc[g] = __builtin_amdgcn_wmma_f32_16x16x32_bf16(
                false, a, false, as_frag(hA[g][kt], hB[g][kt]),
                (short)0, acc[g], false, false);
        }
    }
    // kt=2,3: B-fragments streamed from LDS
#pragma unroll
    for (int kt = 2; kt < 4; ++kt) {
        int da = nn * 64 + kt * 16 + half * 4;
        bf16x16 a = as_frag(*(const uint4*)(sHr + da),
                            *(const uint4*)(sHr + da + 8));
#pragma unroll
        for (int g = 0; g < 4; ++g) {
            int n  = g * 128 + wave * 16 + nn;
            int bi = n * 16 + kt * 4 + half;
            bf16x16 b = as_frag(sWhh[bi], sWhh[bi + 2]);
            acc[g] = __builtin_amdgcn_wmma_f32_16x16x32_bf16(
                false, a, false, b, (short)0, acc[g], false, false);
        }
    }
}

__device__ __forceinline__ void cell_update(f32x8 acc[4], const float bs[4],
                                            f32x8& c, f32x8& h)
{
#pragma unroll
    for (int r = 0; r < 8; ++r) {
        float iv = sigmoid_(acc[0][r] + bs[0]);
        float fv = sigmoid_(acc[1][r] + bs[1]);
        float gv = tanh_(acc[2][r] + bs[2]);
        float ov = sigmoid_(acc[3][r] + bs[3]);
        float cv = fv * c[r] + iv * gv;
        c[r] = cv;
        h[r] = ov * tanh_(cv);
    }
}

// Grid: 64 blocks (one 16-row batch tile each), 256 threads = 8 waves.
// Wave w owns gate columns [16w,16w+16) of all 4 gates -> wave-local i/f/g/o
// nonlinearity and c/h update (c lives in a v8f per lane, accumulator layout).
// Double-buffered h/x staging -> ONE barrier per timestep.  Next-step x is
// global-loaded into registers BEFORE the WMMA block and ds-stored after the
// activations, so its memory latency is hidden under the step's compute.
// Dynamic LDS: W_hh bf16 128KB | h bf16 2x4KB | x_t bf16 2x2KB = 143360 B.
__global__ __launch_bounds__(256)
void lstm_dir_kernel(const unsigned int* __restrict__ xbf, // [64][256][512] dwords
                     const uint4* __restrict__ wihq,       // [512][8] uint4
                     const uint4* __restrict__ whhq,       // [512][16] uint4
                     const float* __restrict__ bsum,       // [512]
                     float* __restrict__ hcatT,            // [256][1024]
                     int nsteps, int rev, int colbase)
{
    extern __shared__ unsigned char smem[];
    uint4*          sWhh = (uint4*)smem;                       // [512][16] 128KB
    unsigned short* sH   = (unsigned short*)(smem + 131072);   // 2 x [16][128]
    unsigned int*   sX   = (unsigned int*)(smem + 139264);     // 2 x [16][32]

    const int tid  = threadIdx.x;
    const int wave = tid >> 5;
    const int lane = tid & 31;
    const int half = lane >> 4;
    const int nn   = lane & 15;

    const unsigned int* xblk = xbf + (size_t)blockIdx.x * TSTEP * 512;

    // W_hh -> LDS (only the kt=2,3 half is read from LDS, but copy is one-time)
    if (nsteps > 1) {
        for (int i = tid; i < 512 * 16; i += 256) sWhh[i] = whhq[i];
    }

    // W_ih B-fragments pinned in registers: [gate][ktile 0..1]  (64 VGPRs)
    uint4 wA[4][2], wB[4][2];
    // W_hh B-fragments for kt=0,1 pinned in registers           (64 VGPRs)
    uint4 hA[4][2], hB[4][2];
#pragma unroll
    for (int g = 0; g < 4; ++g) {
        int n = g * 128 + wave * 16 + nn;
#pragma unroll
        for (int kt = 0; kt < 2; ++kt) {
            int bi = n * 8 + kt * 4 + half;
            wA[g][kt] = wihq[bi];
            wB[g][kt] = wihq[bi + 2];
            int bh = n * 16 + kt * 4 + half;
            hA[g][kt] = whhq[bh];
            hB[g][kt] = whhq[bh + 2];
        }
    }

    float bs[4];
#pragma unroll
    for (int g = 0; g < 4; ++g) bs[g] = bsum[g * 128 + wave * 16 + nn];

    // Stage x for step 0 into buffer 0 (raw 2KB copy, already bf16-swizzled)
    {
        int t0 = rev ? (TSTEP - 1) : 0;
        const uint2* src = (const uint2*)(xblk + (size_t)t0 * 512);
        ((uint2*)sX)[tid] = src[tid];
    }
    __syncthreads();

    f32x8 c = {};
    f32x8 h = {};

    // ---------------- step 0 (peeled; h0 == 0 so no recurrent term) ---------
    {
        uint2 xstage;
        if (nsteps > 1) {   // early-issue the global load for x_{t=1}
            int t1 = rev ? (TSTEP - 2) : 1;
            xstage = ((const uint2*)(xblk + (size_t)t1 * 512))[tid];
        }

        f32x8 acc[4] = {};
        proj_wmma(acc, sX, wA, wB, nn, half);

        cell_update(acc, bs, c, h);

        if (nsteps > 1) {
            ((uint2*)(sX + 512))[tid] = xstage;           // buffer 1
            if (nsteps > 2) {
                int t2 = rev ? (TSTEP - 3) : 2;
                if ((tid & 7) == 0)   // 32 threads x 64B lines cover 2KB
                    __builtin_prefetch(xblk + (size_t)t2 * 512 + (tid >> 3) * 16, 0, 3);
            }
#pragma unroll
            for (int r = 0; r < 8; ++r) {                 // publish h -> buffer 1
                int m = r + 8 * half;
                (sH + 2048)[m * 128 + wave * 16 + nn] = f2bf(h[r]);
            }
        }
        __syncthreads();
    }

    // ---------------- steady state: steps 1 .. nsteps-1 ---------------------
    for (int step = 1; step < nsteps; ++step) {
        const int p = step & 1;
        const unsigned int* sXr = sX + p * 512;
        const unsigned int* sHr = (const unsigned int*)(sH + p * 2048);
        unsigned int*       sXw = sX + (1 - p) * 512;
        unsigned short*     sHw = sH + (1 - p) * 2048;

        // early-issue the global load of x_{t+1} (consumed after activations)
        uint2 xstage;
        const bool more = (step + 1 < nsteps);
        if (more) {
            int t1 = rev ? (TSTEP - 2 - step) : (step + 1);
            xstage = ((const uint2*)(xblk + (size_t)t1 * 512))[tid];
        }

        f32x8 acc[4] = {};
        proj_wmma(acc, sXr, wA, wB, nn, half);
        rec_wmma(acc, sHr, hA, hB, sWhh, wave, nn, half);

        cell_update(acc, bs, c, h);

        if (more) {
            ((uint2*)sXw)[tid] = xstage;
            if (step + 2 < nsteps) {
                int t2 = rev ? (TSTEP - 3 - step) : (step + 2);
                if ((tid & 7) == 0)
                    __builtin_prefetch(xblk + (size_t)t2 * 512 + (tid >> 3) * 16, 0, 3);
            }
        }

#pragma unroll
        for (int r = 0; r < 8; ++r) {   // publish h (bf16, A-matrix layout)
            int m = r + 8 * half;
            sHw[m * 128 + wave * 16 + nn] = f2bf(h[r]);
        }

        __syncthreads();   // single barrier: writes to [1-p] visible next step
    }

    // final hidden -> transposed concat buffer [2H][B]
#pragma unroll
    for (int r = 0; r < 8; ++r) {
        int m = r + 8 * half;
        hcatT[(size_t)(colbase + wave * 16 + nn) * BSZ + blockIdx.x * 16 + m] = h[r];
    }
}

// ---------------------------------------------------------------------------
// Final FC: out[b][o] = sum_k hcatT[k][b] * Wfc[o][k] + bfc[o]
// ---------------------------------------------------------------------------
__global__ __launch_bounds__(256)
void fc_kernel(const float* __restrict__ hcatT, const float* __restrict__ Wfc,
               const float* __restrict__ bfc, float* __restrict__ out)
{
    __shared__ float sw[14 * 256];
    for (int i = threadIdx.x; i < 14 * 256; i += 256) sw[i] = Wfc[i];
    __syncthreads();

    int b = blockIdx.x * 256 + threadIdx.x;
    float acc[14];
#pragma unroll
    for (int o = 0; o < 14; ++o) acc[o] = bfc[o];

    for (int k = 0; k < 256; ++k) {
        float hv = hcatT[(size_t)k * BSZ + b];
#pragma unroll
        for (int o = 0; o < 14; ++o) acc[o] = fmaf(hv, sw[o * 256 + k], acc[o]);
    }
#pragma unroll
    for (int o = 0; o < 14; ++o) out[b * 14 + o] = acc[o];
}

// ---------------------------------------------------------------------------
extern "C" void kernel_launch(void* const* d_in, const int* in_sizes, int n_in,
                              void* d_out, int out_size, void* d_ws, size_t ws_size,
                              hipStream_t stream)
{
    const float* x    = (const float*)d_in[0];
    const float* WihF = (const float*)d_in[1];
    const float* WhhF = (const float*)d_in[2];
    const float* bihF = (const float*)d_in[3];
    const float* bhhF = (const float*)d_in[4];
    const float* WihB = (const float*)d_in[5];
    const float* WhhB = (const float*)d_in[6];
    const float* bihB = (const float*)d_in[7];
    const float* bhhB = (const float*)d_in[8];
    const float* Wfc  = (const float*)d_in[9];
    const float* bfc  = (const float*)d_in[10];
    float* out = (float*)d_out;

    unsigned char* ws = (unsigned char*)d_ws;
    unsigned int* wihFq = (unsigned int*)(ws + 0);        // 64 KB
    unsigned int* whhFq = (unsigned int*)(ws + 65536);    // 128 KB
    unsigned int* wihBq = (unsigned int*)(ws + 196608);   // 64 KB
    unsigned int* whhBq = (unsigned int*)(ws + 262144);   // 128 KB
    float*        bsF   = (float*)(ws + 393216);          // 2 KB
    float*        bsB   = (float*)(ws + 395264);          // 2 KB
    float*        hcatT = (float*)(ws + 397312);          // 1 MB  [256][1024]
    unsigned int* xbf   = (unsigned int*)(ws + 1445888);  // 32 MB [64][256][512]

    prep_x_kernel<<<32768, 256, 0, stream>>>(x, xbf);
    prep_kernel<<<194, 256, 0, stream>>>(WihF, WhhF, bihF, bhhF, wihFq, whhFq, bsF);
    prep_kernel<<<194, 256, 0, stream>>>(WihB, WhhB, bihB, bhhB, wihBq, whhBq, bsB);

    size_t shmem = 131072 + 8192 + 4096;   // 143360 B dynamic LDS (< 320KB WGP)

    // forward: full 256-step recurrence
    lstm_dir_kernel<<<64, 256, shmem, stream>>>(
        xbf, (const uint4*)wihFq, (const uint4*)whhFq, bsF, hcatT, TSTEP, 0, 0);

    // backward direction's contribution to out[:, -1] is exactly ONE cell step
    // on x[:, T-1] from zero state (h_b[:, -1] == first step of reversed scan).
    lstm_dir_kernel<<<64, 256, shmem, stream>>>(
        xbf, (const uint4*)wihBq, (const uint4*)whhBq, bsB, hcatT, 1, 1, 128);

    fc_kernel<<<BSZ / 256, 256, 0, stream>>>(hcatT, Wfc, bfc, out);
}
